// linearized_attention_14628658610495
// MI455X (gfx1250) — matrix operand order repaired and checked by reading, not compiled, not run on hardware
//
#include <hip/hip_runtime.h>
#include <math.h>
#include <stdint.h>

// ---------------- problem constants ----------------
static constexpr int B_   = 16;
static constexpr int T_   = 4096;
static constexpr int CIN  = 128;
static constexpr int D_   = 64;     // C_OUT (head dim)
static constexpr int H_   = 8;
static constexpr int HD   = D_ * H_;   // 512
static constexpr int MP   = 32;        // M=30 padded to 32
static constexpr int MR   = 30;        // real M
static constexpr int SEG  = 8;         // T-segments for kvs accumulation
static constexpr int TSEG = T_ / SEG;  // 512
#define SCL_QK   0.35355339059327373f   /* 64^-0.25 * (1/sqrt(TAU)=1) */
#define RATIO_M  0.18257418583505536f   /* 1/sqrt(30) */
#define EPS_F    1e-6f
#define NEG_BIG  (-3.0e38f)

typedef __attribute__((ext_vector_type(16))) __bf16 v16bf;
typedef __attribute__((ext_vector_type(8)))  float  v8f;

// ---------------- WMMA helpers (CDNA5 layouts) ----------------
__device__ __forceinline__ v8f wmma_bf16(v16bf a, v16bf b, v8f c) {
  return __builtin_amdgcn_wmma_f32_16x16x32_bf16(
      /*neg_a=*/false, a, /*neg_b=*/false, b,
      /*c_mod=*/(short)0, c, /*reuse_a=*/false, /*reuse_b=*/false);
}

// A operand (16x32, bf16) from row-major bf16 [16][ld] at column window kbase.
__device__ __forceinline__ v16bf load_A_rm(const __bf16* base, int ld, int kbase, int lane) {
  int r  = lane & 15;
  int ko = kbase + ((lane >> 4) << 3);
  const __bf16* p = base + r * ld + ko;
  v16bf a;
#pragma unroll
  for (int e = 0; e < 8; ++e) a[e] = p[e];
#pragma unroll
  for (int e = 0; e < 8; ++e) a[8 + e] = p[16 + e];
  return a;
}

// B operand (32x16) with B[k][n] = W[n][k] from PRE-PACKED bf16 row-major W[n][ldw].
__device__ __forceinline__ v16bf load_B_bf(const __bf16* W, int ldw, int n0, int kbase, int lane) {
  int n  = n0 + (lane & 15);
  int kb = kbase + ((lane >> 4) << 3);
  const __bf16* p = W + (size_t)n * ldw + kb;
  v16bf b;
#pragma unroll
  for (int e = 0; e < 8; ++e) b[e] = p[e];
#pragma unroll
  for (int e = 0; e < 8; ++e) b[8 + e] = p[16 + e];
  return b;
}

// B operand from bf16 LDS stored as [k][ldn] (k-major).
__device__ __forceinline__ v16bf load_B_lds(const __bf16* base, int ldn, int n0, int lane) {
  int n  = n0 + (lane & 15);
  int kb = (lane >> 4) << 3;
  v16bf b;
#pragma unroll
  for (int e = 0; e < 8; ++e) b[e] = base[(kb + e) * ldn + n];
#pragma unroll
  for (int e = 0; e < 8; ++e) b[8 + e] = base[(kb + 16 + e) * ldn + n];
  return b;
}

__device__ __forceinline__ void atomicMaxF(float* addr, float val) {
  int* ia = (int*)addr;
  int old = __float_as_int(NEG_BIG);
  while (true) {
    float f = __int_as_float(old);
    if (f >= val) break;
    int assumed = old;
    old = atomicCAS(ia, assumed, __float_as_int(val));
    if (old == assumed) break;
  }
}

// ---------------- kernel 0: pack weights to bf16 + init k-max ----------------
__global__ void k0_pack(const float* __restrict__ wq, const float* __restrict__ wk,
                        const float* __restrict__ wv, const float* __restrict__ wo,
                        const float* __restrict__ proj,
                        __bf16* __restrict__ pwq, __bf16* __restrict__ pwk,
                        __bf16* __restrict__ pwv, __bf16* __restrict__ pwo,
                        __bf16* __restrict__ pproj,
                        float* __restrict__ kmax) {
  int i = blockIdx.x * blockDim.x + threadIdx.x;
  if (i < HD * CIN) {                 // 65536: q/k/v weights
    pwq[i] = (__bf16)wq[i];
    pwk[i] = (__bf16)wk[i];
    pwv[i] = (__bf16)wv[i];
  }
  if (i < D_ * HD) pwo[i] = (__bf16)wo[i];        // 32768
  if (i < MP * D_) {                               // 2048, zero-pad m >= 30
    int m = i >> 6, d = i & 63;
    pproj[i] = (m < MR) ? (__bf16)proj[m * D_ + d] : (__bf16)0.0f;
  }
  if (i < B_ * H_) kmax[i] = NEG_BIG;
}

// ---------------- kernel 1: QKV projection + FAVOR dd/diag + k-max ----------------
// grid: B*(T/16) blocks, 256 threads (8 waves = 8 heads). 16 t-rows per block.
__global__ void __launch_bounds__(256)
k1_proj(const float* __restrict__ x,
        const __bf16* __restrict__ pwq, const float* __restrict__ wqb,
        const __bf16* __restrict__ pwk, const float* __restrict__ wkb,
        const __bf16* __restrict__ pwv, const float* __restrict__ wvb,
        const __bf16* __restrict__ pproj,
        __bf16* __restrict__ v_out,
        float* __restrict__ qdd, float* __restrict__ kdd,
        float* __restrict__ qdiag, float* __restrict__ kdiag,
        float* __restrict__ kmax) {
  __shared__ __bf16 xls[16 * CIN];        // 4 KB: bf16 x tile
  __shared__ __bf16 sls[H_][16 * D_];     // 16 KB: per-wave scaled q/k tile

  const int blk  = blockIdx.x;
  const int b    = blk / (T_ / 16);
  const int t0   = (blk % (T_ / 16)) * 16;
  const int tid  = threadIdx.x;
  const int lane = tid & 31;
  const int h    = tid >> 5;              // wave == head
  const int bh   = b * H_ + h;

  // stage x tile -> LDS (bf16), 8 contiguous floats per thread
  {
    int r = tid >> 4;
    int c = (tid & 15) << 3;
    const float* xp = x + ((size_t)b * T_ + t0 + r) * CIN + c;
#pragma unroll
    for (int e = 0; e < 8; ++e) xls[r * CIN + c + e] = (__bf16)xp[e];
  }
  __syncthreads();

  for (int p = 0; p < 3; ++p) {           // 0:Q 1:K 2:V
    const __bf16* W  = (p == 0) ? pwq : (p == 1) ? pwk : pwv;
    const float*  Wb = (p == 0) ? wqb : (p == 1) ? wkb : wvb;

    v8f acc[4];
#pragma unroll
    for (int nt = 0; nt < 4; ++nt) acc[nt] = v8f{0,0,0,0,0,0,0,0};

#pragma unroll
    for (int ks = 0; ks < 4; ++ks) {      // K = 128 = 4 x 32
      v16bf a = load_A_rm(xls, CIN, ks * 32, lane);
#pragma unroll
      for (int nt = 0; nt < 4; ++nt) {
        v16bf bm = load_B_bf(W, CIN, h * D_ + nt * 16, ks * 32, lane);
        acc[nt] = wmma_bf16(a, bm, acc[nt]);
      }
    }

    if (p == 2) {
      // V: add bias, store bf16 [B][T][512]
#pragma unroll
      for (int nt = 0; nt < 4; ++nt) {
        int col  = h * D_ + nt * 16 + (lane & 15);
        float bi = Wb[col];
#pragma unroll
        for (int i = 0; i < 8; ++i) {
          int row = ((lane >> 4) << 3) + i;
          v_out[((size_t)b * T_ + t0 + row) * HD + col] = (__bf16)(acc[nt][i] + bi);
        }
      }
    } else {
      // Q/K: scale, stash tile in LDS for the dd GEMM
#pragma unroll
      for (int nt = 0; nt < 4; ++nt) {
        int cl   = nt * 16 + (lane & 15);
        float bi = Wb[h * D_ + cl];
#pragma unroll
        for (int i = 0; i < 8; ++i) {
          int row = ((lane >> 4) << 3) + i;
          sls[h][row * D_ + cl] = (__bf16)((acc[nt][i] + bi) * SCL_QK);
        }
      }
      __syncthreads();

      // diag = 0.5 * sum_d scaled^2 (one row per lane 0..15)
      if (lane < 16) {
        float dg = 0.f;
#pragma unroll 8
        for (int d = 0; d < D_; ++d) { float v = (float)sls[h][lane * D_ + d]; dg += v * v; }
        float* dgp = (p == 0) ? qdiag : kdiag;
        dgp[(size_t)bh * T_ + t0 + lane] = 0.5f * dg;
      }

      // dd = scaled @ proj^T  (16 x 32, K = 64 -> 2 steps, 2 n-tiles)
      v8f dacc[2] = { v8f{0,0,0,0,0,0,0,0}, v8f{0,0,0,0,0,0,0,0} };
#pragma unroll
      for (int ks = 0; ks < 2; ++ks) {
        v16bf a2 = load_A_rm(sls[h], D_, ks * 32, lane);
#pragma unroll
        for (int nt = 0; nt < 2; ++nt) {
          v16bf bp = load_B_bf(pproj, D_, nt * 16, ks * 32, lane);
          dacc[nt] = wmma_bf16(a2, bp, dacc[nt]);
        }
      }

      float* dd = ((p == 0) ? qdd : kdd) + ((size_t)bh * T_ + t0) * MP;
      float tmax = NEG_BIG;
#pragma unroll
      for (int nt = 0; nt < 2; ++nt) {
        int m = nt * 16 + (lane & 15);
#pragma unroll
        for (int i = 0; i < 8; ++i) {
          int row = ((lane >> 4) << 3) + i;
          float val = dacc[nt][i];
          dd[(size_t)row * MP + m] = val;
          if (m < MR) tmax = fmaxf(tmax, val);
        }
      }
      if (p == 1) {                        // global (t,m) max per (b,h)
        tmax = fmaxf(tmax, __shfl_xor(tmax, 1, 32));
        tmax = fmaxf(tmax, __shfl_xor(tmax, 2, 32));
        tmax = fmaxf(tmax, __shfl_xor(tmax, 4, 32));
        tmax = fmaxf(tmax, __shfl_xor(tmax, 8, 32));
        tmax = fmaxf(tmax, __shfl_xor(tmax, 16, 32));
        if (lane == 0) atomicMaxF(&kmax[bh], tmax);
      }
      __syncthreads();
    }
  }
}

// ---------------- kernel 2: partial kvs = phi_k^T @ V over one T-segment ----------------
// grid: B*H*SEG blocks, 256 threads (8 waves -> 2 m-tiles x 4 d-tiles).
__global__ void __launch_bounds__(256)
k2_kvs(const __bf16* __restrict__ v_in,
       const float* __restrict__ kdd, const float* __restrict__ kdiag,
       const float* __restrict__ kmaxp,
       float* __restrict__ pkvs, float* __restrict__ pksum) {
  __shared__ __bf16 phils[32 * MP];   // [t][m]  2 KB
  __shared__ __bf16 vls[32 * D_];     // [t][d]  4 KB

  const int blk  = blockIdx.x;
  const int bh   = blk / SEG;
  const int seg  = blk % SEG;
  const int b    = bh >> 3;
  const int h    = bh & 7;
  const int tid  = threadIdx.x;
  const int lane = tid & 31;
  const int w    = tid >> 5;
  const int mt   = w >> 2;            // m-tile 0..1
  const int nt   = w & 3;             // d-tile 0..3

  const float kmx = kmaxp[bh];
  const float* ddp = kdd   + (size_t)bh * T_ * MP;
  const float* dgp = kdiag + (size_t)bh * T_;

  v8f acc = v8f{0,0,0,0,0,0,0,0};
  float ks_acc = 0.f;                  // threads 0..31 accumulate ksum[m=tid]

  const int tbeg = seg * TSEG, tend = tbeg + TSEG;
  for (int tc = tbeg; tc < tend; tc += 32) {
    // prefetch next chunk of kdd while this one is consumed (global_prefetch_b8)
    if (tc + 32 < tend) {
      const char* pf = (const char*)(ddp + (size_t)(tc + 32) * MP);
      __builtin_prefetch(pf + tid * 16, 0, 1);                    // 4 KB of kdd
    }
    __syncthreads();
    // V chunk via CDNA5 async global->LDS DMA: one B128 per thread = 4 KB
    {
      int tl = tid >> 3;
      int d0 = (tid & 7) << 3;
      const __bf16* vp = v_in + ((size_t)b * T_ + tc + tl) * HD + h * D_ + d0;
      unsigned lds_off = (unsigned)(uintptr_t)&vls[tl * D_ + d0];
      asm volatile("global_load_async_to_lds_b128 %0, %1, off"
                   :: "v"(lds_off), "v"(vp) : "memory");
    }
    // phi_k chunk: 32x32 values, 4 per thread (padded m -> 0)
#pragma unroll
    for (int j = 0; j < 4; ++j) {
      int idx = tid * 4 + j;
      int tl  = idx >> 5, m = idx & 31;
      float ph = 0.f;
      if (m < MR) {
        float dd = ddp[(size_t)(tc + tl) * MP + m];
        ph = RATIO_M * (__expf(dd - dgp[tc + tl] - kmx) + EPS_F);
      }
      phils[idx] = (__bf16)ph;
    }
    asm volatile("s_wait_asynccnt 0" ::: "memory");
    __syncthreads();

    if (tid < 32) {                    // ksum[m=tid] partial
#pragma unroll 8
      for (int tl = 0; tl < 32; ++tl) ks_acc += (float)phils[tl * MP + tid];
    }

    // A = phi^T (rows = m, K = t) read k-major from phils[t][m]
    v16bf a;
    {
      int m  = mt * 16 + (lane & 15);
      int ko = (lane >> 4) << 3;
#pragma unroll
      for (int e = 0; e < 8; ++e) a[e]     = phils[(ko + e) * MP + m];
#pragma unroll
      for (int e = 0; e < 8; ++e) a[8 + e] = phils[(ko + 16 + e) * MP + m];
    }
    v16bf bm = load_B_lds(vls, D_, nt * 16, lane);
    acc = wmma_bf16(a, bm, acc);
  }

  // write partial kvs[blk][m][d] (f32)
  {
    int d = nt * 16 + (lane & 15);
#pragma unroll
    for (int i = 0; i < 8; ++i) {
      int m = mt * 16 + ((lane >> 4) << 3) + i;
      pkvs[((size_t)blk * MP + m) * D_ + d] = acc[i];
    }
  }
  if (tid < 32) pksum[(size_t)blk * MP + tid] = ks_acc;
}

// ---------------- kernel 2r: reduce SEG partials -> kvs, ksum ----------------
__global__ void __launch_bounds__(256)
k2_reduce(const float* __restrict__ pkvs, const float* __restrict__ pksum,
          float* __restrict__ kvs, float* __restrict__ ksum) {
  const int bh  = blockIdx.x;
  const int tid = threadIdx.x;
  for (int i = tid; i < MP * D_; i += 256) {
    float s = 0.f;
#pragma unroll
    for (int g = 0; g < SEG; ++g)
      s += pkvs[((size_t)(bh * SEG + g) * MP) * D_ + i];
    kvs[(size_t)bh * MP * D_ + i] = s;
  }
  if (tid < MP) {
    float s = 0.f;
#pragma unroll
    for (int g = 0; g < SEG; ++g)
      s += pksum[(size_t)(bh * SEG + g) * MP + tid];
    ksum[(size_t)bh * MP + tid] = s;
  }
}

// ---------------- kernel 3: phi_q, num/den, output projection ----------------
// grid: B*(T/16) blocks, 256 threads (wave h = head h).
__global__ void __launch_bounds__(256)
k3_out(const float* __restrict__ qdd, const float* __restrict__ qdiag,
       const float* __restrict__ kvs, const float* __restrict__ ksum,
       const __bf16* __restrict__ pwo, const float* __restrict__ wob,
       float* __restrict__ out) {
  __shared__ __bf16 kvls[H_][MP * D_];  // 32 KB
  __shared__ float  ksls[H_ * MP];      // 1 KB
  __shared__ float  denls[H_][16];
  __shared__ __bf16 yls[16 * HD];       // 16 KB

  const int blk  = blockIdx.x;
  const int b    = blk / (T_ / 16);
  const int t0   = (blk % (T_ / 16)) * 16;
  const int tid  = threadIdx.x;
  const int lane = tid & 31;
  const int h    = tid >> 5;
  const int bh   = b * H_ + h;

  // stage this head's kvs (bf16) + ksum into LDS
  {
    const float* kp = kvs + (size_t)bh * MP * D_;
    for (int i = lane; i < MP * D_; i += 32) kvls[h][i] = (__bf16)kp[i];
    ksls[h * MP + lane] = ksum[(size_t)bh * MP + lane];
  }
  __syncthreads();

  // phi_q directly in A layout: lane half-rows; row-max via shfl_xor(16)
  const float* ddp = qdd + ((size_t)bh * T_ + t0) * MP;
  const int r  = lane & 15;
  const int ko = (lane >> 4) << 3;
  float ph[16];
#pragma unroll
  for (int e = 0; e < 8; ++e) ph[e]     = ddp[(size_t)r * MP + ko + e];
#pragma unroll
  for (int e = 0; e < 8; ++e) ph[8 + e] = ddp[(size_t)r * MP + ko + 16 + e];

  float mx = NEG_BIG;
#pragma unroll
  for (int e = 0; e < 16; ++e) {
    int m = (e < 8) ? (ko + e) : (ko + 16 + e - 8);
    if (m < MR) mx = fmaxf(mx, ph[e]);
  }
  mx = fmaxf(mx, __shfl_xor(mx, 16, 32));
  const float dg = qdiag[(size_t)bh * T_ + t0 + r];

  v16bf a;
  float den = 0.f;
#pragma unroll
  for (int e = 0; e < 16; ++e) {
    int m = (e < 8) ? (ko + e) : (ko + 16 + e - 8);
    float p = (m < MR) ? RATIO_M * (__expf(ph[e] - dg - mx) + EPS_F) : 0.f;
    a[e] = (__bf16)p;
    den += p * ksls[h * MP + m];
  }
  den += __shfl_xor(den, 16, 32);
  if (lane < 16) denls[h][lane] = den;

  // num = phi_q @ kvs  (K = 32, one WMMA per d-tile)
  v8f nacc[4];
#pragma unroll
  for (int nt = 0; nt < 4; ++nt) {
    v16bf bm = load_B_lds(kvls[h], D_, nt * 16, lane);
    nacc[nt] = wmma_bf16(a, bm, v8f{0,0,0,0,0,0,0,0});
  }
  __syncthreads();

  // y = num / den -> bf16 LDS [16][512]
#pragma unroll
  for (int nt = 0; nt < 4; ++nt) {
    int dcol = nt * 16 + (lane & 15);
#pragma unroll
    for (int i = 0; i < 8; ++i) {
      int row = ((lane >> 4) << 3) + i;
      yls[row * HD + h * D_ + dcol] = (__bf16)(nacc[nt][i] / denls[h][row]);
    }
  }
  __syncthreads();

  // out = y @ wo^T + wo_b   (16 x 64, K = 512 -> 16 steps; waves 0..3)
  if (h < 4) {
    v8f oacc = v8f{0,0,0,0,0,0,0,0};
#pragma unroll
    for (int ks = 0; ks < 16; ++ks) {
      v16bf a2 = load_A_rm(yls, HD, ks * 32, lane);
      v16bf b2 = load_B_bf(pwo, HD, h * 16, ks * 32, lane);
      oacc = wmma_bf16(a2, b2, oacc);
    }
    int col  = h * 16 + (lane & 15);
    float bi = wob[col];
    float* op = out + ((size_t)b * T_ + t0) * D_;
#pragma unroll
    for (int i = 0; i < 8; ++i) {
      int row = ((lane >> 4) << 3) + i;
      op[(size_t)row * D_ + col] = oacc[i] + bi;
    }
  }
}

// ---------------- host launcher ----------------
extern "C" void kernel_launch(void* const* d_in, const int* in_sizes, int n_in,
                              void* d_out, int out_size, void* d_ws, size_t ws_size,
                              hipStream_t stream) {
  const float* x    = (const float*)d_in[0];
  const float* wq_w = (const float*)d_in[1];
  const float* wq_b = (const float*)d_in[2];
  const float* wk_w = (const float*)d_in[3];
  const float* wk_b = (const float*)d_in[4];
  const float* wv_w = (const float*)d_in[5];
  const float* wv_b = (const float*)d_in[6];
  const float* wo_w = (const float*)d_in[7];
  const float* wo_b = (const float*)d_in[8];
  const float* proj = (const float*)d_in[9];

  // workspace layout (bytes)
  char* ws = (char*)d_ws;
  const size_t SZ_V    = (size_t)B_ * T_ * HD * 2;            // bf16 V:  64 MB
  const size_t SZ_DD   = (size_t)B_ * H_ * T_ * MP * 4;       // f32 dd:  64 MB each
  const size_t SZ_DIAG = (size_t)B_ * H_ * T_ * 4;            //           2 MB each
  const size_t SZ_KMAX = 512;
  const size_t SZ_KVS  = (size_t)B_ * H_ * MP * D_ * 4;       //           1 MB
  const size_t SZ_KSUM = (size_t)B_ * H_ * MP * 4;
  const size_t SZ_PKVS = (size_t)B_ * H_ * SEG * MP * D_ * 4; //           8 MB
  const size_t SZ_PKSM = (size_t)B_ * H_ * SEG * MP * 4;
  const size_t SZ_PW   = (size_t)HD * CIN * 2;                // packed qkv weight
  const size_t SZ_PWO  = (size_t)D_ * HD * 2;
  const size_t SZ_PPRJ = (size_t)MP * D_ * 2;
  size_t off = 0;
  __bf16* v_bf   = (__bf16*)(ws + off); off += SZ_V;
  float*  qdd    = (float*) (ws + off); off += SZ_DD;
  float*  kdd    = (float*) (ws + off); off += SZ_DD;
  float*  qdiag  = (float*) (ws + off); off += SZ_DIAG;
  float*  kdiag  = (float*) (ws + off); off += SZ_DIAG;
  float*  kmax   = (float*) (ws + off); off += SZ_KMAX;
  float*  kvs    = (float*) (ws + off); off += SZ_KVS;
  float*  ksum   = (float*) (ws + off); off += SZ_KSUM;
  float*  pkvs   = (float*) (ws + off); off += SZ_PKVS;
  float*  pksum  = (float*) (ws + off); off += SZ_PKSM;
  __bf16* pwq    = (__bf16*)(ws + off); off += SZ_PW;
  __bf16* pwk    = (__bf16*)(ws + off); off += SZ_PW;
  __bf16* pwv    = (__bf16*)(ws + off); off += SZ_PW;
  __bf16* pwo    = (__bf16*)(ws + off); off += SZ_PWO;
  __bf16* pproj  = (__bf16*)(ws + off); off += SZ_PPRJ;

  float* out = (float*)d_out;
  const int nTileBlocks = B_ * (T_ / 16);   // 4096

  k0_pack<<<(HD * CIN + 255) / 256, 256, 0, stream>>>(wq_w, wk_w, wv_w, wo_w, proj,
                                                      pwq, pwk, pwv, pwo, pproj, kmax);
  k1_proj<<<nTileBlocks, 256, 0, stream>>>(x, pwq, wq_b, pwk, wk_b, pwv, wv_b,
                                           pproj, v_bf, qdd, kdd, qdiag, kdiag, kmax);
  k2_kvs<<<B_ * H_ * SEG, 256, 0, stream>>>(v_bf, kdd, kdiag, kmax, pkvs, pksum);
  k2_reduce<<<B_ * H_, 256, 0, stream>>>(pkvs, pksum, kvs, ksum);
  k3_out<<<nTileBlocks, 256, 0, stream>>>(qdd, qdiag, kvs, ksum, pwo, wo_b, out);
}
